// MoELayer_2886218023254
// MI455X (gfx1250) — matrix operand order repaired and checked by reading, not compiled, not run on hardware
//
#include <hip/hip_runtime.h>

#define MODEL_D 1024
#define HID     2048
#define NEXP    8
#define BATCH_N 8192
#define CAP     8192            // worst-case tokens per expert (top-2 distinct)
#define MT      32              // tokens per FFN block
#define NCHUNK  256             // N columns per chunk
#define KSTEP   32              // K per WMMA step (bf16 16x16x32)
#define MAXTILES (CAP / MT)     // 256
#define XPAD    8               // Xs/Hs row pad: stride mod 64 dwords == 4
#define WPAD    8               // slab row pad: 40 halfwords = 20 dwords

typedef __attribute__((ext_vector_type(16))) __bf16 v16bf;
typedef __attribute__((ext_vector_type(8)))  float  v8f;
typedef __attribute__((ext_vector_type(4)))  int    v4i;

union FragBF { uint4 u[2]; v16bf v; };

typedef unsigned short SlabRow[KSTEP + WPAD];

// ---- CDNA5 async Global->LDS copy (ASYNCcnt) with portable fallback ----
#if defined(__has_builtin)
#  if __has_builtin(__builtin_amdgcn_global_load_async_to_lds_b128) && \
      __has_builtin(__builtin_amdgcn_s_wait_asynccnt)
#    define USE_ASYNC_COPY 1
#  endif
#endif
#ifndef USE_ASYNC_COPY
#  define USE_ASYNC_COPY 0
#endif

#if USE_ASYNC_COPY
typedef __attribute__((address_space(1))) v4i* gv4i_p;   // global int4*
typedef __attribute__((address_space(3))) v4i* lv4i_p;   // LDS int4*
#endif

__device__ __forceinline__ unsigned short f2bf(float f) {
  unsigned u = __float_as_uint(f);
  u += 0x7FFFu + ((u >> 16) & 1u);   // round-to-nearest-even
  return (unsigned short)(u >> 16);
}

// Copy this thread's 64B slab row (32 bf16) from pre-transposed weights to LDS.
__device__ __forceinline__ void stage_slab(SlabRow* buf, const unsigned short* g,
                                           int tid) {
#if USE_ASYNC_COPY
#pragma unroll
  for (int c = 0; c < 4; ++c)
    __builtin_amdgcn_global_load_async_to_lds_b128(
        (gv4i_p)(g + c * 8), (lv4i_p)(&buf[tid][c * 8]), 0, 0);
#else
#pragma unroll
  for (int c = 0; c < 4; ++c)
    *(uint4*)(&buf[tid][c * 8]) = *(const uint4*)(g + c * 8);
#endif
}

__device__ __forceinline__ void wait_stage() {
#if USE_ASYNC_COPY
  __builtin_amdgcn_s_wait_asynccnt(0);
#endif
}

// ---------------------------------------------------------------------------
// Zero moe_out accumulator + per-expert counters
// ---------------------------------------------------------------------------
__global__ void moe_zero_kernel(float* __restrict__ out, int n, int* __restrict__ cnt) {
  int i = blockIdx.x * blockDim.x + threadIdx.x;
  if (i < n) out[i] = 0.f;
  if (i < NEXP) cnt[i] = 0;
}

// ---------------------------------------------------------------------------
// One-shot fp32 [E][K][N] -> bf16 [E][N][K] transpose-convert (64x64 LDS tiles)
// ---------------------------------------------------------------------------
__global__ void __launch_bounds__(256) moe_cvt_tr_kernel(
    const float* __restrict__ src, unsigned short* __restrict__ dst, int K, int N) {
  __shared__ unsigned short T[64][64 + 8];
  const int tid = threadIdx.x;
  const int kb = blockIdx.y * 64, nb = blockIdx.x * 64;
  const float* s = src + ((size_t)blockIdx.z * K + kb) * N + nb;
#pragma unroll
  for (int j = 0; j < 4; ++j) {
    int idx = j * 256 + tid;            // 0..1023
    int r  = idx >> 4;                  // k-row 0..63
    int c4 = (idx & 15) * 4;            // n-col 0..60
    float4 v = *(const float4*)(s + (size_t)r * N + c4);
    T[c4 + 0][r] = f2bf(v.x); T[c4 + 1][r] = f2bf(v.y);
    T[c4 + 2][r] = f2bf(v.z); T[c4 + 3][r] = f2bf(v.w);
  }
  __syncthreads();
  unsigned short* d = dst + ((size_t)blockIdx.z * N + nb) * K + kb;
#pragma unroll
  for (int j = 0; j < 2; ++j) {
    int idx = j * 256 + tid;            // 0..511
    int r  = idx >> 3;                  // n-row 0..63
    int c8 = (idx & 7) * 8;             // k-col 0..56
    *(uint4*)(d + (size_t)r * K + c8) = *(const uint4*)&T[r][c8];
  }
}

// ---------------------------------------------------------------------------
// Gating: one wave32 per token. logits = x@Wg, top-2 softmax, bucket scatter.
// ---------------------------------------------------------------------------
__global__ void __launch_bounds__(256) moe_gate_kernel(
    const float* __restrict__ x, const float* __restrict__ Wg,
    float* __restrict__ logits, int* __restrict__ topidx,
    int* __restrict__ cnt, int* __restrict__ bucket, float* __restrict__ gwt) {
  const int lane  = threadIdx.x & 31;
  const int token = blockIdx.x * 8 + (threadIdx.x >> 5);

  float acc[NEXP];
#pragma unroll
  for (int e = 0; e < NEXP; ++e) acc[e] = 0.f;

  const float* xr = x + (size_t)token * MODEL_D;
  for (int i = 0; i < MODEL_D / 32; ++i) {
    float xv = xr[lane + 32 * i];
    const float* wr = Wg + (size_t)(lane + 32 * i) * NEXP;
#pragma unroll
    for (int e = 0; e < NEXP; ++e) acc[e] += xv * wr[e];
  }
#pragma unroll
  for (int off = 16; off > 0; off >>= 1) {
#pragma unroll
    for (int e = 0; e < NEXP; ++e) acc[e] += __shfl_down(acc[e], off, 32);
  }
  if (lane == 0) {
#pragma unroll
    for (int e = 0; e < NEXP; ++e) logits[(size_t)token * NEXP + e] = acc[e];
    int i0 = 0; float v0 = acc[0];
#pragma unroll
    for (int e = 1; e < NEXP; ++e) if (acc[e] > v0) { v0 = acc[e]; i0 = e; }
    int i1 = (i0 == 0) ? 1 : 0; float v1 = acc[i1];
#pragma unroll
    for (int e = 0; e < NEXP; ++e) if (e != i0 && acc[e] > v1) { v1 = acc[e]; i1 = e; }
    float e1  = __expf(v1 - v0);
    float inv = 1.f / (1.f + e1);
    topidx[token * 2 + 0] = i0;
    topidx[token * 2 + 1] = i1;
    int p0 = atomicAdd(cnt + i0, 1);
    bucket[i0 * CAP + p0] = token; gwt[i0 * CAP + p0] = inv;
    int p1 = atomicAdd(cnt + i1, 1);
    bucket[i1 * CAP + p1] = token; gwt[i1 * CAP + p1] = e1 * inv;
  }
}

// ---------------------------------------------------------------------------
// Grouped dual-GEMM FFN per (expert, 32-token tile), bf16 WMMA,
// double-buffered async weight slabs from pre-transposed bf16 weights.
// ---------------------------------------------------------------------------
__global__ void __launch_bounds__(256) moe_ffn_kernel(
    const float* __restrict__ x,
    const unsigned short* __restrict__ W1T,   // bf16 [E][HID][MODEL_D]
    const float* __restrict__ b1,
    const unsigned short* __restrict__ W2T,   // bf16 [E][MODEL_D][HID]
    const float* __restrict__ b2,
    const int* __restrict__ cnt, const int* __restrict__ bucket,
    const float* __restrict__ gwt, float* __restrict__ out) {
  __shared__ __align__(16) unsigned short Xs[MT][MODEL_D + XPAD]; //  66,048 B
  __shared__ __align__(16) unsigned short Hs[MT][HID + XPAD];     // 131,584 B
  __shared__ __align__(16) SlabRow Wd[2][NCHUNK];                 //  40,960 B
  __shared__ int   toks[MT];
  __shared__ float gws[MT];

  const int e     = blockIdx.x >> 8;
  const int tile  = blockIdx.x & (MAXTILES - 1);
  const int count = cnt[e];
  if (tile * MT >= count) return;            // block-uniform early exit

  const int tid  = threadIdx.x;
  const int lane = tid & 31;
  const int wave = tid >> 5;
  const int mrow = (wave & 1) * 16;          // wave's M half of the token tile
  const int nsub = wave >> 1;                // wave's N quartet within a chunk

  // ---- stage token ids / gates ----
  if (tid < MT) {
    int r = tile * MT + tid;
    int tok = 0; float g = 0.f;
    if (r < count) { tok = bucket[e * CAP + r]; g = gwt[e * CAP + r]; }
    toks[tid] = tok; gws[tid] = g;
  }
  __syncthreads();

  // ---- stage X tile (fp32 -> bf16), coalesced float4 loads ----
#pragma unroll
  for (int j = 0; j < 4; ++j) {
    int r = j * 8 + wave;
    const float* xr = x + (size_t)toks[r] * MODEL_D;
#pragma unroll
    for (int c = 0; c < 8; ++c) {
      int col = c * 128 + lane * 4;
      float4 v = *(const float4*)(xr + col);
      unsigned short* p = &Xs[r][col];
      p[0] = f2bf(v.x); p[1] = f2bf(v.y); p[2] = f2bf(v.z); p[3] = f2bf(v.w);
    }
  }
  __syncthreads();

  const unsigned short* W1e = W1T + (size_t)e * HID * MODEL_D;
  const unsigned short* W2e = W2T + (size_t)e * MODEL_D * HID;

  // ================= Stage A: H = relu(X @ W1 + b1) =================
  for (int nc = 0; nc < HID; nc += NCHUNK) {
    const unsigned short* wrow = W1e + (size_t)(nc + tid) * MODEL_D;
    v8f acc[4] = {};
    stage_slab(Wd[0], wrow, tid);
    wait_stage();
    __syncthreads();
    int cur = 0;
    for (int kk = 0; kk < MODEL_D; kk += KSTEP) {
      if (kk + KSTEP < MODEL_D)
        stage_slab(Wd[cur ^ 1], wrow + kk + KSTEP, tid);   // prefetch next slab
      FragBF fa;                               // A: row=lane&15, K per ISA layout
      const unsigned short* ap = &Xs[mrow + (lane & 15)][kk + 8 * (lane >> 4)];
      fa.u[0] = *(const uint4*)ap;
      fa.u[1] = *(const uint4*)(ap + 16);
#pragma unroll
      for (int q = 0; q < 4; ++q) {
        FragBF fb;                             // B: col=lane&15, K per ISA layout
        const unsigned short* bp = &Wd[cur][(nsub * 4 + q) * 16 + (lane & 15)][(lane >> 4) * 8];
        fb.u[0] = *(const uint4*)bp;
        fb.u[1] = *(const uint4*)(bp + 16);
        acc[q] = __builtin_amdgcn_wmma_f32_16x16x32_bf16(
            false, fa.v, false, fb.v, (short)0, acc[q], false, false);
      }
      wait_stage();
      __syncthreads();
      cur ^= 1;
    }
    // bias + relu -> bf16 H tile in LDS
#pragma unroll
    for (int q = 0; q < 4; ++q) {
      int col = nc + (nsub * 4 + q) * 16 + (lane & 15);
      float bias = b1[(size_t)e * HID + col];
#pragma unroll
      for (int i = 0; i < 8; ++i) {
        int r = mrow + i + 8 * (lane >> 4);
        float v = acc[q][i] + bias;
        Hs[r][col] = f2bf(v > 0.f ? v : 0.f);
      }
    }
  }
  __syncthreads();

  // ================= Stage B: out += gate * (H @ W2 + b2) =================
  for (int nc = 0; nc < MODEL_D; nc += NCHUNK) {
    const unsigned short* wrow = W2e + (size_t)(nc + tid) * HID;
    v8f acc[4] = {};
    stage_slab(Wd[0], wrow, tid);
    wait_stage();
    __syncthreads();
    int cur = 0;
    for (int kk = 0; kk < HID; kk += KSTEP) {
      if (kk + KSTEP < HID)
        stage_slab(Wd[cur ^ 1], wrow + kk + KSTEP, tid);
      FragBF fa;
      const unsigned short* ap = &Hs[mrow + (lane & 15)][kk + 8 * (lane >> 4)];
      fa.u[0] = *(const uint4*)ap;
      fa.u[1] = *(const uint4*)(ap + 16);
#pragma unroll
      for (int q = 0; q < 4; ++q) {
        FragBF fb;
        const unsigned short* bp = &Wd[cur][(nsub * 4 + q) * 16 + (lane & 15)][(lane >> 4) * 8];
        fb.u[0] = *(const uint4*)bp;
        fb.u[1] = *(const uint4*)(bp + 16);
        acc[q] = __builtin_amdgcn_wmma_f32_16x16x32_bf16(
            false, fa.v, false, fb.v, (short)0, acc[q], false, false);
      }
      wait_stage();
      __syncthreads();
      cur ^= 1;
    }
    // gate-scale + scatter-add (token appears in two expert buckets)
#pragma unroll
    for (int q = 0; q < 4; ++q) {
      int col = nc + (nsub * 4 + q) * 16 + (lane & 15);
      float bias2 = b2[(size_t)e * MODEL_D + col];
#pragma unroll
      for (int i = 0; i < 8; ++i) {
        int r = mrow + i + 8 * (lane >> 4);
        float v = (acc[q][i] + bias2) * gws[r];
        atomicAdd(out + (size_t)toks[r] * MODEL_D + col, v);
      }
    }
  }
}

// ---------------------------------------------------------------------------
extern "C" void kernel_launch(void* const* d_in, const int* in_sizes, int n_in,
                              void* d_out, int out_size, void* d_ws, size_t ws_size,
                              hipStream_t stream) {
  (void)in_sizes; (void)n_in; (void)out_size; (void)ws_size;
  const float* x  = (const float*)d_in[0];
  const float* Wg = (const float*)d_in[1];
  const float* W1 = (const float*)d_in[2];
  const float* b1 = (const float*)d_in[3];
  const float* W2 = (const float*)d_in[4];
  const float* b2 = (const float*)d_in[5];

  float* out_moe    = (float*)d_out;                              // [8192,1024]
  float* out_logits = out_moe + (size_t)BATCH_N * MODEL_D;        // [8192,8]
  int*   out_idx    = (int*)(out_logits + (size_t)BATCH_N * NEXP);// [8192,2]

  // workspace layout (all 256B-aligned offsets)
  char*  ws     = (char*)d_ws;
  int*   cnt    = (int*)ws;                                       // 8 ints
  int*   bucket = (int*)(ws + 256);                               // 8*8192 ints
  float* gwt    = (float*)(ws + 256 + (size_t)NEXP * CAP * 4);    // 8*8192 f32
  size_t off    = 256 + (size_t)NEXP * CAP * 8;                   // = 524544
  unsigned short* W1T = (unsigned short*)(ws + off);              // 32 MB bf16
  unsigned short* W2T = W1T + (size_t)NEXP * HID * MODEL_D;       // 32 MB bf16

  int nz = BATCH_N * MODEL_D;
  moe_zero_kernel<<<(nz + 255) / 256, 256, 0, stream>>>(out_moe, nz, cnt);
  // W1 [E][1024][2048] -> W1T [E][2048][1024]; W2 [E][2048][1024] -> W2T [E][1024][2048]
  moe_cvt_tr_kernel<<<dim3(HID / 64, MODEL_D / 64, NEXP), 256, 0, stream>>>(
      W1, W1T, MODEL_D, HID);
  moe_cvt_tr_kernel<<<dim3(MODEL_D / 64, HID / 64, NEXP), 256, 0, stream>>>(
      W2, W2T, HID, MODEL_D);
  moe_gate_kernel<<<BATCH_N / 8, 256, 0, stream>>>(x, Wg, out_logits, out_idx,
                                                   cnt, bucket, gwt);
  moe_ffn_kernel<<<NEXP * MAXTILES, 256, 0, stream>>>(x, W1T, b1, W2T, b2,
                                                      cnt, bucket, gwt, out_moe);
}
